// ShallowPLRNN_7112465842752
// MI455X (gfx1250) — matrix operand order repaired and checked by reading, not compiled, not run on hardware
//
#include <hip/hip_runtime.h>
#include <hip/hip_bf16.h>

typedef __bf16 bf16;
typedef __attribute__((ext_vector_type(16))) __bf16 v16bf;
typedef __attribute__((ext_vector_type(8)))  float  v8f;

#define BATCH  256
#define LATENT 256
#define HIDDEN 1024
#define OBS    256

union BV { uint4 u[2]; v16bf v; };

// A-operand (16x32 bf16, M x K) from row-major [16][stride] buffer.
// Lane layout (ISA 7.12.2): lanes 0-15 K = {0..7,16..23}, lanes 16-31 K = {8..15,24..31}.
__device__ __forceinline__ v16bf load_a_tile(const bf16* base, int stride, int kbase, int lane) {
    int m    = lane & 15;
    int koff = (lane & 16) ? 8 : 0;
    const uint4* p = reinterpret_cast<const uint4*>(base + m * stride + kbase + koff);
    BV r;
    r.u[0] = p[0];   // K = koff .. koff+7
    r.u[1] = p[2];   // K = koff+16 .. koff+23
    return r.v;
}

// B-operand (32x16 bf16, K x N) for C = X @ W^T : B[k][n] = W[n][k].
// Lane n%16 holds K = {0..7,16..23} of row n of W (lanes 16-31: K = {8..15,24..31}).
__device__ __forceinline__ v16bf load_b_tile(const bf16* w, int ldk, int nbase, int kbase, int lane) {
    int n  = nbase + (lane & 15);
    int kb = kbase + ((lane & 16) ? 8 : 0);
    const uint4* p = reinterpret_cast<const uint4*>(w + (size_t)n * ldk + kb);
    BV r;
    r.u[0] = p[0];
    r.u[1] = p[2];
    return r.v;
}

__device__ __forceinline__ v8f wmma_bf16(v16bf a, v16bf b, v8f c) {
    return __builtin_amdgcn_wmma_f32_16x16x32_bf16(false, a, false, b, (short)0, c, false, false);
}

__global__ void convert_f32_bf16(const float* __restrict__ src, bf16* __restrict__ dst, int n) {
    int i = blockIdx.x * blockDim.x + threadIdx.x;
    if (i < n) dst[i] = (bf16)src[i];
}

__global__ __launch_bounds__(256)
void plrnn_scan_kernel(const float* __restrict__ z0,
                       const int*   __restrict__ seq_len_p,
                       const bf16*  __restrict__ Abf,
                       const bf16*  __restrict__ W1bf,
                       const bf16*  __restrict__ W2bf,
                       const float* __restrict__ h1,
                       const float* __restrict__ h2,
                       const bf16*  __restrict__ Bbf,
                       const float* __restrict__ bvec,
                       float*       __restrict__ out)
{
    __shared__ bf16 zbuf[2][16][LATENT];   // double-buffered z tile (bf16)
    __shared__ bf16 hbuf[16][HIDDEN];      // hidden activations (bf16)

    const int tid   = threadIdx.x;
    const int lane  = tid & 31;
    const int wv    = tid >> 5;            // wave id 0..7
    const int mbase = blockIdx.x * 16;     // batch-row tile base
    const int T     = *seq_len_p;

    // Load this block's 16 rows of z0 -> bf16 LDS
    for (int i = tid; i < 16 * LATENT; i += 256) {
        int r = i >> 8, c = i & (LATENT - 1);
        zbuf[0][r][c] = (bf16)z0[(size_t)(mbase + r) * LATENT + c];
    }
    __syncthreads();

    const int mb = (lane >> 4) << 3;   // accumulator row base: 0 or 8
    const int nl = lane & 15;          // accumulator column within tile

    for (int t = 0; t < T; ++t) {
        const int cur = t & 1, nxt = cur ^ 1;
        const bf16* zc = &zbuf[cur][0][0];

        // ---- Phase 1: hidden = relu(z @ W2^T + h2)   [16 x 1024] ----
        // (z A-tiles get hoisted by the compiler across the 8 n-tiles)
        for (int i = 0; i < 8; ++i) {
            int ntile = wv * 8 + i;                 // 64 n-tiles over 8 waves
            v8f acc = {};
            #pragma unroll
            for (int kt = 0; kt < LATENT / 32; ++kt) {
                v16bf a = load_a_tile(zc, LATENT, kt * 32, lane);
                v16bf b = load_b_tile(W2bf, LATENT, ntile * 16, kt * 32, lane);
                acc = wmma_bf16(a, b, acc);
            }
            float bias = h2[ntile * 16 + nl];
            #pragma unroll
            for (int r = 0; r < 8; ++r)
                hbuf[mb + r][ntile * 16 + nl] = (bf16)fmaxf(acc[r] + bias, 0.0f);
        }
        __syncthreads();

        // ---- Phase 2: z_next = z @ A^T + hidden @ W1^T + h1   [16 x 256] ----
        // k-outer / n-inner with two accumulators: each A tile read once,
        // feeds two independent WMMA chains (halves LDS reads, hides RAW).
        {
            const int n0 = (wv * 2) * 16, n1 = (wv * 2 + 1) * 16;
            v8f acc0 = {}, acc1 = {};
            #pragma unroll
            for (int kt = 0; kt < LATENT / 32; ++kt) {
                v16bf a  = load_a_tile(zc, LATENT, kt * 32, lane);
                v16bf b0 = load_b_tile(Abf, LATENT, n0, kt * 32, lane);
                v16bf b1 = load_b_tile(Abf, LATENT, n1, kt * 32, lane);
                acc0 = wmma_bf16(a, b0, acc0);
                acc1 = wmma_bf16(a, b1, acc1);
            }
            #pragma unroll 4
            for (int kt = 0; kt < HIDDEN / 32; ++kt) {
                v16bf a  = load_a_tile(&hbuf[0][0], HIDDEN, kt * 32, lane);
                v16bf b0 = load_b_tile(W1bf, HIDDEN, n0, kt * 32, lane);
                v16bf b1 = load_b_tile(W1bf, HIDDEN, n1, kt * 32, lane);
                acc0 = wmma_bf16(a, b0, acc0);
                acc1 = wmma_bf16(a, b1, acc1);
            }
            float bias0 = h1[n0 + nl];
            float bias1 = h1[n1 + nl];
            #pragma unroll
            for (int r = 0; r < 8; ++r) {
                zbuf[nxt][mb + r][n0 + nl] = (bf16)(acc0[r] + bias0);
                zbuf[nxt][mb + r][n1 + nl] = (bf16)(acc1[r] + bias1);
            }
        }
        __syncthreads();

        // ---- Phase 3: x_t = z_next @ B^T + b  -> out[m][t][n] ----
        {
            const int n0 = (wv * 2) * 16, n1 = (wv * 2 + 1) * 16;
            v8f acc0 = {}, acc1 = {};
            #pragma unroll
            for (int kt = 0; kt < LATENT / 32; ++kt) {
                v16bf a  = load_a_tile(&zbuf[nxt][0][0], LATENT, kt * 32, lane);
                v16bf b0 = load_b_tile(Bbf, LATENT, n0, kt * 32, lane);
                v16bf b1 = load_b_tile(Bbf, LATENT, n1, kt * 32, lane);
                acc0 = wmma_bf16(a, b0, acc0);
                acc1 = wmma_bf16(a, b1, acc1);
            }
            float bias0 = bvec[n0 + nl];
            float bias1 = bvec[n1 + nl];
            // Non-temporal stores: output is write-once streaming (134 MB);
            // keep the 192MB L2 reserved for the weights all 16 WGs re-read
            // every timestep.
            #pragma unroll
            for (int r = 0; r < 8; ++r) {
                size_t row = (size_t)(mbase + mb + r) * T + t;
                __builtin_nontemporal_store(acc0[r] + bias0, &out[row * OBS + n0 + nl]);
                __builtin_nontemporal_store(acc1[r] + bias1, &out[row * OBS + n1 + nl]);
            }
        }
        __syncthreads();
    }
}

extern "C" void kernel_launch(void* const* d_in, const int* in_sizes, int n_in,
                              void* d_out, int out_size, void* d_ws, size_t ws_size,
                              hipStream_t stream) {
    // setup_inputs order: z0, seq_len, A, W1, W2, h1, h2, B, b
    const float* z0   = (const float*)d_in[0];
    const int*   slen = (const int*)  d_in[1];
    const float* A    = (const float*)d_in[2];
    const float* W1   = (const float*)d_in[3];
    const float* W2   = (const float*)d_in[4];
    const float* h1   = (const float*)d_in[5];
    const float* h2   = (const float*)d_in[6];
    const float* B    = (const float*)d_in[7];
    const float* bv   = (const float*)d_in[8];
    float* out = (float*)d_out;

    // Workspace layout (bf16 weight copies, ~1.31 MB total)
    char* ws = (char*)d_ws;
    bf16* Abf  = (bf16*)(ws);                                    // 256*256
    bf16* W1bf = (bf16*)(ws + 131072);                           // 256*1024
    bf16* W2bf = (bf16*)(ws + 131072 + 524288);                  // 1024*256
    bf16* Bbf  = (bf16*)(ws + 131072 + 524288 + 524288);         // 256*256

    const int nA  = LATENT * LATENT;
    const int nW1 = LATENT * HIDDEN;
    const int nW2 = HIDDEN * LATENT;
    const int nB  = OBS * LATENT;
    convert_f32_bf16<<<(nA  + 255) / 256, 256, 0, stream>>>(A,  Abf,  nA);
    convert_f32_bf16<<<(nW1 + 255) / 256, 256, 0, stream>>>(W1, W1bf, nW1);
    convert_f32_bf16<<<(nW2 + 255) / 256, 256, 0, stream>>>(W2, W2bf, nW2);
    convert_f32_bf16<<<(nB  + 255) / 256, 256, 0, stream>>>(B,  Bbf,  nB);

    // One workgroup per 16-row batch tile; scan the whole sequence in-kernel.
    plrnn_scan_kernel<<<BATCH / 16, 256, 0, stream>>>(
        z0, slen, Abf, W1bf, W2bf, h1, h2, Bbf, bv, out);
}
